// FGEncodedGAT_33303176413852
// MI455X (gfx1250) — matrix operand order repaired and checked by reading, not compiled, not run on hardware
//
#include <hip/hip_runtime.h>
#include <math.h>

typedef __attribute__((ext_vector_type(16))) __bf16 v16bf;
typedef __attribute__((ext_vector_type(8)))  float  v8f;

#define DEV __device__ __forceinline__

DEV unsigned short f32_to_bf16_bits(float f) {
  unsigned int u = __float_as_uint(f);
  u += 0x7FFFu + ((u >> 16) & 1u);          // round-to-nearest-even
  return (unsigned short)(u >> 16);
}
DEV unsigned int pack_bf16x2(float lo, float hi) {
  return (unsigned int)f32_to_bf16_bits(lo) | ((unsigned int)f32_to_bf16_bits(hi) << 16);
}
DEV float elu1(float x) { return x > 0.f ? x : (__expf(x) - 1.f); }
DEV void atomicMaxFloat(float* a, float v) {
  if (v >= 0.f) atomicMax((int*)a, __float_as_int(v));
  else          atomicMin((unsigned int*)a, __float_as_uint(v));
}

union FragU { v16bf v; uint4 q[2]; };

DEV v16bf load_frag2(const unsigned short* p0, const unsigned short* p1) {
  FragU u;
  u.q[0] = *(const uint4*)p0;
  u.q[1] = *(const uint4*)p1;
  return u.v;
}
DEV v8f wmma_bf16(v16bf a, v16bf b, v8f c) {
  return __builtin_amdgcn_wmma_f32_16x16x32_bf16(false, a, false, b, (short)0, c, false, false);
}

// ============ WMMA GEMM: C[M,N] = act(A[M,K] @ W[K,N] + bias[N]) ============
// block = 256 threads (8 wave32) arranged 4x2; wave tile 32x32 -> block tile 128x64.
// BK = 64. Requires: K % 64 == 0, N % 64 == 0 (true for every GEMM in this net).
__global__ __launch_bounds__(256) void gemm_bf16_wmma(
    const float* __restrict__ A, const float* __restrict__ W,
    const float* __restrict__ bias, float* __restrict__ C,
    int M, int N, int K, int act)
{
  __shared__ __align__(16) unsigned short As[128 * 72];  // 128 x 64 (stride 72 = 144B)
  __shared__ __align__(16) unsigned short Ws[64 * 72];   // transposed [n][k]: 64 x 64

  const int tid  = threadIdx.x;
  const int wave = tid >> 5, lane = tid & 31;
  const int wr   = wave >> 1;          // 0..3 row group
  const int wc   = wave & 1;           // 0..1 col group
  const int row0 = blockIdx.y * 128;
  const int col0 = blockIdx.x * 64;

  // cooperative-load coordinates
  const int arow_ld = tid >> 4;   // 0..15 (A: 16 float4 per 64-wide row)
  const int ac4     = tid & 15;
  const int wr_ld   = tid >> 4;   // W: 64x64 floats; per i: row k = i*16 + (tid>>4)
  const int wc4     = tid & 15;

  v8f acc00 = {0.f,0.f,0.f,0.f,0.f,0.f,0.f,0.f};
  v8f acc01 = {0.f,0.f,0.f,0.f,0.f,0.f,0.f,0.f};
  v8f acc10 = {0.f,0.f,0.f,0.f,0.f,0.f,0.f,0.f};
  v8f acc11 = {0.f,0.f,0.f,0.f,0.f,0.f,0.f,0.f};

  for (int k0 = 0; k0 < K; k0 += 64) {
    // ---- stage global -> regs, branchless ragged-M (clamp + mask) ----
    float4 a_reg[8];
    #pragma unroll
    for (int i = 0; i < 8; ++i) {
      int gr  = row0 + i * 16 + arow_ld;
      int grc = (gr < M) ? gr : (M - 1);
      float4 v = *(const float4*)(A + (size_t)grc * K + k0 + ac4 * 4);
      float msk = (gr < M) ? 1.f : 0.f;
      v.x *= msk; v.y *= msk; v.z *= msk; v.w *= msk;
      a_reg[i] = v;
    }
    float4 w_reg[4];
    #pragma unroll
    for (int i = 0; i < 4; ++i) {
      int r = i * 16 + wr_ld;          // k index 0..63
      w_reg[i] = *(const float4*)(W + (size_t)(k0 + r) * N + col0 + wc4 * 4);
    }
    if (k0 + 64 < K)
      __builtin_prefetch(A + (size_t)(row0 + arow_ld) * K + k0 + 64, 0, 2);

    __syncthreads();   // previous iteration's LDS reads complete

    // ---- regs -> LDS ----
    #pragma unroll
    for (int i = 0; i < 8; ++i) {
      int r = i * 16 + arow_ld;
      *(uint2*)&As[r * 72 + ac4 * 4] =
          make_uint2(pack_bf16x2(a_reg[i].x, a_reg[i].y),
                     pack_bf16x2(a_reg[i].z, a_reg[i].w));
    }
    #pragma unroll
    for (int i = 0; i < 4; ++i) {
      int r = i * 16 + wr_ld;          // k index within tile
      Ws[(wc4 * 4 + 0) * 72 + r] = f32_to_bf16_bits(w_reg[i].x);
      Ws[(wc4 * 4 + 1) * 72 + r] = f32_to_bf16_bits(w_reg[i].y);
      Ws[(wc4 * 4 + 2) * 72 + r] = f32_to_bf16_bits(w_reg[i].z);
      Ws[(wc4 * 4 + 3) * 72 + r] = f32_to_bf16_bits(w_reg[i].w);
    }
    __syncthreads();

    // ---- fragments + 8 WMMAs (2x2 per k-chunk, full operand reuse) ----
    const int kba = (lane >> 4) * 8;    // A: halves 0-7 -> k=kba+h, 8-15 -> k=kba+16+h'
    const int kbb = (lane >> 4) * 8;    // B: same as A: 0-7 -> k=kbb+h, 8-15 -> k=kbb+16+h'
    const int ar0 = wr * 32 + (lane & 15);
    const int bn0 = wc * 32 + (lane & 15);
    #pragma unroll
    for (int kc = 0; kc < 2; ++kc) {
      const int ko = kc * 32;
      v16bf a0 = load_frag2(&As[ar0 * 72 + ko + kba],
                            &As[ar0 * 72 + ko + kba + 16]);
      v16bf a1 = load_frag2(&As[(ar0 + 16) * 72 + ko + kba],
                            &As[(ar0 + 16) * 72 + ko + kba + 16]);
      v16bf b0 = load_frag2(&Ws[bn0 * 72 + ko + kbb],
                            &Ws[bn0 * 72 + ko + kbb + 16]);
      v16bf b1 = load_frag2(&Ws[(bn0 + 16) * 72 + ko + kbb],
                            &Ws[(bn0 + 16) * 72 + ko + kbb + 16]);
      acc00 = wmma_bf16(a0, b0, acc00);
      acc01 = wmma_bf16(a0, b1, acc01);
      acc10 = wmma_bf16(a1, b0, acc10);
      acc11 = wmma_bf16(a1, b1, acc11);
    }
    __syncthreads();
  }

  // ---- epilogue: C/D layout lane -> n=lane&15, m=(lane>=16?8:0)+vgpr ----
  const int n  = lane & 15;
  const int mb = (lane >> 4) * 8;
  const int cA = col0 + wc * 32 + n;
  const int cB = cA + 16;
  const float bvA = bias ? bias[cA] : 0.f;
  const float bvB = bias ? bias[cB] : 0.f;
  const int rbase0 = row0 + wr * 32 + mb;
  #pragma unroll
  for (int r = 0; r < 8; ++r) {
    int g0 = rbase0 + r;        // tile row set 0
    int g1 = g0 + 16;           // tile row set 1
    if (g0 < M) {
      float v0 = acc00[r] + bvA, v1 = acc01[r] + bvB;
      if (act == 1) { v0 = elu1(v0); v1 = elu1(v1); }
      C[(size_t)g0 * N + cA] = v0;
      C[(size_t)g0 * N + cB] = v1;
    }
    if (g1 < M) {
      float v0 = acc10[r] + bvA, v1 = acc11[r] + bvB;
      if (act == 1) { v0 = elu1(v0); v1 = elu1(v1); }
      C[(size_t)g1 * N + cA] = v0;
      C[(size_t)g1 * N + cB] = v1;
    }
  }
}

// ================= element / fill kernels =================
__global__ void fill_k(float* __restrict__ p, float v, long long n) {
  long long i = (long long)blockIdx.x * blockDim.x + threadIdx.x;
  if (i < n) p[i] = v;
}

// ================= GATv2 edge kernels =================
__global__ void edge_logits_k(
    const float* __restrict__ xl, const float* __restrict__ xr,
    const float* __restrict__ ea, const float* __restrict__ We,
    const float* __restrict__ att, const int* __restrict__ src,
    const int* __restrict__ dst, float* __restrict__ logits,
    float* __restrict__ maxb, int E, int H, int Cc)
{
  int t = blockIdx.x * blockDim.x + threadIdx.x;
  if (t >= E * H) return;
  int e = t / H, h = t - e * H;
  int s = src[e], d = dst[e];
  float ef[6];
  #pragma unroll
  for (int k = 0; k < 6; ++k) ef[k] = ea[(size_t)e * 6 + k];
  const int HC = H * Cc, HC4 = HC / 4;
  const float4* xl4 = (const float4*)(xl + (size_t)s * HC + h * Cc);
  const float4* xr4 = (const float4*)(xr + (size_t)d * HC + h * Cc);
  const float4* av4 = (const float4*)(att + h * Cc);
  const float4* Wb4 = (const float4*)(We + h * Cc);
  float accv = 0.f;
  for (int c4 = 0; c4 < Cc / 4; ++c4) {
    float4 w0 = Wb4[c4],           w1 = Wb4[HC4 + c4],     w2 = Wb4[2 * HC4 + c4];
    float4 w3 = Wb4[3 * HC4 + c4], w4 = Wb4[4 * HC4 + c4], w5 = Wb4[5 * HC4 + c4];
    float4 xa = xl4[c4], xb = xr4[c4], at = av4[c4];
    #define GAT_COMP(f) { \
      float m = xa.f + xb.f + (ef[0]*w0.f + ef[1]*w1.f + ef[2]*w2.f \
                             + ef[3]*w3.f + ef[4]*w4.f + ef[5]*w5.f); \
      m = (m > 0.f) ? m : 0.2f * m; \
      accv = __fmaf_rn(at.f, m, accv); }
    GAT_COMP(x) GAT_COMP(y) GAT_COMP(z) GAT_COMP(w)
    #undef GAT_COMP
  }
  logits[t] = accv;
  atomicMaxFloat(&maxb[(size_t)d * H + h], accv);
}

__global__ void edge_exp_k(const int* __restrict__ dst, float* __restrict__ logits,
                           const float* __restrict__ maxb, float* __restrict__ sumb,
                           int E, int H)
{
  int t = blockIdx.x * blockDim.x + threadIdx.x;
  if (t >= E * H) return;
  int e = t / H, h = t - e * H;
  int d = dst[e];
  float m = maxb[(size_t)d * H + h];
  if (!__builtin_isfinite(m)) m = 0.f;
  float ex = __expf(logits[t] - m);
  logits[t] = ex;
  atomicAdd(&sumb[(size_t)d * H + h], ex);
}

__global__ void edge_scatter_k(const float* __restrict__ xl, const int* __restrict__ src,
                               const int* __restrict__ dst, const float* __restrict__ ex,
                               const float* __restrict__ sumb, float* __restrict__ accb,
                               int E, int H, int Cc)
{
  int t = blockIdx.x * blockDim.x + threadIdx.x;
  if (t >= E * H) return;
  int e = t / H, h = t - e * H;
  int s = src[e], d = dst[e];
  const int HC = H * Cc;
  float alpha = ex[t] / (sumb[(size_t)d * H + h] + 1e-16f);
  const float4* xls4 = (const float4*)(xl + (size_t)s * HC + h * Cc);
  float* op = accb + (size_t)d * HC + h * Cc;
  for (int c4 = 0; c4 < Cc / 4; ++c4) {
    float4 v = xls4[c4];
    atomicAdd(&op[c4 * 4 + 0], alpha * v.x);
    atomicAdd(&op[c4 * 4 + 1], alpha * v.y);
    atomicAdd(&op[c4 * 4 + 2], alpha * v.z);
    atomicAdd(&op[c4 * 4 + 3], alpha * v.w);
  }
}

__global__ void node_bias_elu_k(float* __restrict__ a, const float* __restrict__ bias,
                                long long n, int D)
{
  long long i = (long long)blockIdx.x * blockDim.x + threadIdx.x;
  if (i >= n) return;
  int j = (int)(i % D);
  a[i] = elu1(a[i] + bias[j]);
}

// ================= pooling / FG helpers =================
__global__ void pool_k(const float* __restrict__ h, const int* __restrict__ seg,
                       float* __restrict__ g, long long n, int D)
{
  long long i = (long long)blockIdx.x * blockDim.x + threadIdx.x;
  if (i >= n) return;
  long long node = i / D; int j = (int)(i % D);
  atomicAdd(&g[(size_t)seg[node] * D + j], h[i]);
}

__global__ void gather_emb_k(const float* __restrict__ emb, const int* __restrict__ ids,
                             float* __restrict__ o, long long n)
{
  long long i = (long long)blockIdx.x * blockDim.x + threadIdx.x;
  if (i >= n) return;
  long long r = i / 64; int j = (int)(i % 64);
  o[i] = emb[(size_t)ids[r] * 64 + j];
}

__global__ void a2f_scatter_k(const float* __restrict__ msg, const int* __restrict__ ai,
                              const int* __restrict__ fi, float* __restrict__ num,
                              long long n)
{
  long long i = (long long)blockIdx.x * blockDim.x + threadIdx.x;
  if (i >= n) return;
  long long t = i / 128; int j = (int)(i % 128);
  atomicAdd(&num[(size_t)fi[t] * 128 + j], msg[(size_t)ai[t] * 128 + j]);
}

__global__ void a2f_cnt_k(const int* __restrict__ fi, float* __restrict__ cnt, int n)
{
  int i = blockIdx.x * blockDim.x + threadIdx.x;
  if (i < n) atomicAdd(&cnt[fi[i]], 1.f);
}

__global__ void combine_mean_k(float* __restrict__ fgx, const float* __restrict__ num,
                               const float* __restrict__ cnt, long long n)
{
  long long i = (long long)blockIdx.x * blockDim.x + threadIdx.x;
  if (i >= n) return;
  long long r = i / 128;
  fgx[i] += num[i] / fmaxf(cnt[r], 1.f);
}

__global__ void concat_k(const float* __restrict__ ga, const float* __restrict__ gf,
                         float* __restrict__ o, long long n)
{
  long long i = (long long)blockIdx.x * blockDim.x + threadIdx.x;
  if (i >= n) return;
  long long b = i / 256; int j = (int)(i % 256);
  o[i] = (j < 128) ? ga[b * 128 + j] : gf[b * 128 + (j - 128)];
}

__global__ void out_k(const float* __restrict__ g, const float* __restrict__ Wo,
                      const float* __restrict__ bo, float* __restrict__ y, int B)
{
  int b = blockIdx.x * blockDim.x + threadIdx.x;
  if (b >= B) return;
  float acc = bo[0];
  for (int c = 0; c < 128; ++c) acc = __fmaf_rn(g[(size_t)b * 128 + c], Wo[c], acc);
  y[b] = acc;
}

// ================= host side =================
struct GatP { const float *Wl, *bl, *Wr, *br, *We, *att, *bias; };

extern "C" void kernel_launch(void* const* d_in, const int* in_sizes, int n_in,
                              void* d_out, int out_size, void* d_ws, size_t ws_size,
                              hipStream_t stream) {
  (void)in_sizes; (void)n_in; (void)out_size; (void)ws_size;
  int ii = 0;
  auto F = [&]() { return (const float*)d_in[ii++]; };
  auto I = [&]() { return (const int*)d_in[ii++]; };
  auto G = [&]() { GatP g; g.Wl=F(); g.bl=F(); g.Wr=F(); g.br=F(); g.We=F(); g.att=F(); g.bias=F(); return g; };

  // setup_inputs() dict insertion order, params nested in insertion order
  const float* x            = F();
  const float* edge_attr    = F();
  const float* fg_edge_attr = F();
  GatP ac0 = G(), ac1 = G(), ac2 = G();
  const float* afcW = F();  const float* afcB = F();
  const float* emb  = F();
  const float* ftpW = F();  const float* ftpB = F();
  const float* a2fW = F();  const float* a2fB = F();
  GatP fc0 = G(), fc1 = G();
  const float* fgfcW = F(); const float* fgfcB = F();
  const float* fuseW = F(); const float* fuseB = F();
  const float* outW  = F(); const float* outB  = F();
  const int* ei       = I();
  const int* batch    = I();
  const int* fg_ids   = I();
  const int* fei      = I();
  const int* a2f      = I();
  const int* fg_batch = I();

  const int N = 50000, E = 200000, NFG = 15000, EFG = 50000, A2F = 50000, B = 1024;
  const int *a_src = ei,  *a_dst = ei + E;
  const int *f_src = fei, *f_dst = fei + EFG;
  const int *a2f_a = a2f, *a2f_f = a2f + A2F;

  float* ws = (float*)d_ws;
  size_t off = 0;
  auto alloc = [&](size_t n) { float* p = ws + off; off += n; return p; };
  float* P     = alloc((size_t)N * 512);   // layer input / accumulator
  float* Q     = alloc((size_t)N * 512);   // xl
  float* R     = alloc((size_t)N * 512);   // xr (dead after logits)
  float* LOGB  = alloc((size_t)E * 4);
  float* MAXB  = alloc((size_t)N * 4);
  float* SUMB  = alloc((size_t)N * 4);
  float* GAin  = alloc((size_t)B * 128);
  float* GAout = alloc((size_t)B * 128);
  float* GFin  = alloc((size_t)B * 128);
  float* GFout = alloc((size_t)B * 128);
  float* GCAT  = alloc((size_t)B * 256);
  float* GFUSE = alloc((size_t)B * 128);
  float* FGEMB = alloc((size_t)NFG * 64);
  float* NUM   = alloc((size_t)NFG * 128);
  float* CNT   = alloc((size_t)NFG);

  auto fill = [&](float* p, size_t n, float v) {
    fill_k<<<(unsigned)((n + 255) / 256), 256, 0, stream>>>(p, v, (long long)n);
  };
  auto gemm = [&](const float* A, const float* W, const float* bias, float* C,
                  int M, int Nn, int K, int act) {
    dim3 grid((unsigned)(Nn / 64), (unsigned)((M + 127) / 128));
    gemm_bf16_wmma<<<grid, 256, 0, stream>>>(A, W, bias, C, M, Nn, K, act);
  };
  auto gat = [&](const float* hin, int Nn, int Kin, int H, int Cc,
                 const int* src, const int* dst, int Ee, const float* ea,
                 const GatP& p, float* xl, float* xr, float* accb) {
    const int HC = H * Cc;
    gemm(hin, p.Wl, p.bl, xl, Nn, HC, Kin, 0);
    gemm(hin, p.Wr, p.br, xr, Nn, HC, Kin, 0);
    fill(MAXB, (size_t)Nn * H, -INFINITY);
    fill(SUMB, (size_t)Nn * H, 0.f);
    const int nt = Ee * H;
    const unsigned ge = (unsigned)((nt + 255) / 256);
    edge_logits_k<<<ge, 256, 0, stream>>>(xl, xr, ea, p.We, p.att, src, dst,
                                          LOGB, MAXB, Ee, H, Cc);
    edge_exp_k<<<ge, 256, 0, stream>>>(dst, LOGB, MAXB, SUMB, Ee, H);
    fill(accb, (size_t)Nn * HC, 0.f);   // hin may alias accb: all its readers queued before
    edge_scatter_k<<<ge, 256, 0, stream>>>(xl, src, dst, LOGB, SUMB, accb, Ee, H, Cc);
    const size_t ne = (size_t)Nn * HC;
    node_bias_elu_k<<<(unsigned)((ne + 255) / 256), 256, 0, stream>>>(
        accb, p.bias, (long long)ne, HC);
  };

  // ---- atom branch: 3 GATv2 layers (rotate through P) ----
  gat(x, N, 64,  4, 128, a_src, a_dst, E, edge_attr, ac0, Q, R, P);
  gat(P, N, 512, 4, 128, a_src, a_dst, E, edge_attr, ac1, Q, R, P);
  gat(P, N, 512, 1, 128, a_src, a_dst, E, edge_attr, ac2, Q, R, P);
  fill(GAin, (size_t)B * 128, 0.f);
  { size_t nt = (size_t)N * 128;
    pool_k<<<(unsigned)((nt + 255) / 256), 256, 0, stream>>>(P, batch, GAin, (long long)nt, 128); }
  gemm(GAin, afcW, afcB, GAout, B, 128, 128, 1);   // elu fused

  // ---- functional-group branch ----
  { size_t nt = (size_t)NFG * 64;
    gather_emb_k<<<(unsigned)((nt + 255) / 256), 256, 0, stream>>>(emb, fg_ids, FGEMB, (long long)nt); }
  gemm(FGEMB, ftpW, ftpB, P, NFG, 128, 64, 0);     // fg_x base -> P
  gemm(x, a2fW, a2fB, Q, N, 128, 64, 0);           // atom_msg -> Q
  fill(NUM, (size_t)NFG * 128, 0.f);
  fill(CNT, (size_t)NFG, 0.f);
  { size_t nt = (size_t)A2F * 128;
    a2f_scatter_k<<<(unsigned)((nt + 255) / 256), 256, 0, stream>>>(Q, a2f_a, a2f_f, NUM, (long long)nt); }
  a2f_cnt_k<<<(A2F + 255) / 256, 256, 0, stream>>>(a2f_f, CNT, A2F);
  { size_t nt = (size_t)NFG * 128;
    combine_mean_k<<<(unsigned)((nt + 255) / 256), 256, 0, stream>>>(P, NUM, CNT, (long long)nt); }

  gat(P, NFG, 128, 4, 128, f_src, f_dst, EFG, fg_edge_attr, fc0, Q, R, P);
  gat(P, NFG, 512, 1, 128, f_src, f_dst, EFG, fg_edge_attr, fc1, Q, R, P);
  fill(GFin, (size_t)B * 128, 0.f);
  { size_t nt = (size_t)NFG * 128;
    pool_k<<<(unsigned)((nt + 255) / 256), 256, 0, stream>>>(P, fg_batch, GFin, (long long)nt, 128); }
  gemm(GFin, fgfcW, fgfcB, GFout, B, 128, 128, 1); // elu fused

  // ---- fuse + readout ----
  { size_t nt = (size_t)B * 256;
    concat_k<<<(unsigned)((nt + 255) / 256), 256, 0, stream>>>(GAout, GFout, GCAT, (long long)nt); }
  gemm(GCAT, fuseW, fuseB, GFUSE, B, 128, 256, 1); // elu fused
  out_k<<<(B + 255) / 256, 256, 0, stream>>>(GFUSE, outW, outB, (float*)d_out, B);
}
